// InceptionResNet_11373073400148
// MI455X (gfx1250) — matrix operand order repaired and checked by reading, not compiled, not run on hardware
//
#include <hip/hip_runtime.h>
#include <hip/hip_bf16.h>

// ---------------------------------------------------------------------------
// Sparse-voxel InceptionResNet block on gfx1250 (MI455X), bf16 WMMA pipeline.
// N=262144 voxels, C=128 ch, K=27 offsets. Memory/gather bound; bf16
// activations keep the 27x gather working set (67MB) resident in the 192MB L2.
// Weights are pre-swizzled to the WMMA B lane layout and double-buffered into
// LDS with async global->LDS copies (ASYNCcnt); gather rows are prefetched one
// kernel offset ahead.
// ---------------------------------------------------------------------------

typedef unsigned int  u32;
typedef unsigned short u16;
typedef __attribute__((ext_vector_type(4)))  u32    u32x4;
typedef __attribute__((ext_vector_type(8)))  float  v8f;
typedef __attribute__((ext_vector_type(16))) __bf16 v16bf;
typedef int v4i_vs __attribute__((vector_size(16)));   // builtin's pointee type

#define NVOX 262144
#define KOFF 27

#ifndef __has_builtin
#define __has_builtin(x) 0
#endif

#if __has_builtin(__builtin_amdgcn_global_load_async_to_lds_b128)
#define HAVE_ASYNC_LDS 1
#else
#define HAVE_ASYNC_LDS 0
#endif

struct FragPair { u32x4 lo, hi; };

static __device__ __forceinline__ v16bf make_frag(u32x4 lo, u32x4 hi) {
  FragPair p{lo, hi};
  return __builtin_bit_cast(v16bf, p);
}

static __device__ __forceinline__ u16 f2bf(float f) {
  u32 u = __builtin_bit_cast(u32, f);
  u32 r = u + 0x7FFFu + ((u >> 16) & 1u);   // round-to-nearest-even
  return (u16)(r >> 16);
}
static __device__ __forceinline__ u32 pack2(float a, float b) {
  return (u32)f2bf(a) | ((u32)f2bf(b) << 16);
}

static __device__ __forceinline__ v8f wmma_bf16(v16bf a, v16bf b, v8f c) {
  return __builtin_amdgcn_wmma_f32_16x16x32_bf16(false, a, false, b, (short)0, c,
                                                 false, false);
}

// 16B async copy global->LDS (gfx1250 GLOBAL_LOAD_ASYNC_TO_LDS_B128).
static __device__ __forceinline__ void async_copy16(const void* gsrc, void* ldst) {
#if HAVE_ASYNC_LDS
  __builtin_amdgcn_global_load_async_to_lds_b128(
      (__attribute__((address_space(1))) v4i_vs*)(gsrc),
      (__attribute__((address_space(3))) v4i_vs*)(ldst), 0, 0);
#endif
}

// Fence for staged LDS data: drain this wave's async copies, then barrier.
static __device__ __forceinline__ void stage_fence() {
#if HAVE_ASYNC_LDS
#if __has_builtin(__builtin_amdgcn_s_wait_asynccnt)
  __builtin_amdgcn_s_wait_asynccnt(0);
#else
  asm volatile("s_wait_asynccnt 0x0" ::: "memory");
#endif
#endif
  __syncthreads();
}

// ---------------------------------------------------------------------------
// Preprocessing kernels
// ---------------------------------------------------------------------------

// x [N,128] f32 -> xb [N,128] bf16 (packed u32 pairs); also zero the dummy row.
__global__ __launch_bounds__(256) void k_convert_x(const float2* __restrict__ x,
                                                   u32* __restrict__ xb,
                                                   u32* __restrict__ zerorow) {
  int gid = blockIdx.x * blockDim.x + threadIdx.x;
  if (gid < 128) zerorow[gid] = 0u;
  int n = NVOX * 64;
  for (int id = gid; id < n; id += gridDim.x * blockDim.x) {
    float2 v = x[id];
    xb[id] = pack2(v.x, v.y);
  }
}

// nbr [N,27] + mask [N,27] (bool bytes) -> cidx [27,N] with -1 for masked.
__global__ __launch_bounds__(256) void k_build_cidx(const int* __restrict__ nbr,
                                                    const unsigned char* __restrict__ mask,
                                                    int* __restrict__ cidx) {
  int n = KOFF * NVOX;
  for (int id = blockIdx.x * blockDim.x + threadIdx.x; id < n;
       id += gridDim.x * blockDim.x) {
    int k = id / NVOX;
    int i = id - k * NVOX;
    int src = i * KOFF + k;
    cidx[id] = mask[src] ? nbr[src] : -1;
  }
}

// Pre-swizzle an f32 weight [numOff, Kdim, Ndim] into the WMMA bf16 B-matrix
// lane layout: pk[(((off*Kt + kt)*Nt + nt)*32 + lane)*8 + j] packs rows
// (32*kt + 16*(j>>2) + 8*(lane>>4) + 2*(j&3), +1) of column (nt*16 + (lane&15)). A B fragment
// is then exactly 2 consecutive b128 loads per lane.
__global__ __launch_bounds__(256) void k_pack_b(const float* __restrict__ W,
                                                u32* __restrict__ pk,
                                                int numOff, int Kdim, int Ndim) {
  int Kt = Kdim >> 5, Nt = Ndim >> 4;
  int total = numOff * Kt * Nt * 256;
  for (int id = blockIdx.x * blockDim.x + threadIdx.x; id < total;
       id += gridDim.x * blockDim.x) {
    int j    = id & 7;
    int lane = (id >> 3) & 31;
    int nt   = (id >> 8) % Nt;
    int kt   = ((id >> 8) / Nt) % Kt;
    int off  = (id >> 8) / (Nt * Kt);
    int col  = nt * 16 + (lane & 15);
    int krow = 32 * kt + 16 * (j >> 2) + 8 * (lane >> 4) + 2 * (j & 3);
    const float* Wb = W + (size_t)off * Kdim * Ndim;
    pk[id] = pack2(Wb[(size_t)krow * Ndim + col],
                   Wb[(size_t)(krow + 1) * Ndim + col]);
  }
}

// ---------------------------------------------------------------------------
// Stage 1: u0 = relu(x @ W10 + b10)   [N,32] bf16.   K=128 -> 4 kt, 2 nt.
// ---------------------------------------------------------------------------
__global__ __launch_bounds__(256) void k_u0(const char* __restrict__ xb,
                                            const u32* __restrict__ w10p,
                                            const float* __restrict__ b10,
                                            u16* __restrict__ u0) {
  int tid = threadIdx.x, lane = tid & 31, wave = tid >> 5, hi = lane >> 4;
  int rowbase = blockIdx.x * 128 + wave * 16;
  const char* xrow = xb + (size_t)(rowbase + (lane & 15)) * 256;
  const u32x4* bq = (const u32x4*)w10p;
  v8f acc0 = {0,0,0,0,0,0,0,0}, acc1 = acc0;
#pragma unroll
  for (int kt = 0; kt < 4; ++kt) {
    u32x4 alo = *(const u32x4*)(xrow + kt * 64 + hi * 16);
    u32x4 ahi = *(const u32x4*)(xrow + kt * 64 + hi * 16 + 32);
    v16bf a = make_frag(alo, ahi);
    int f0 = ((kt * 2 + 0) * 32 + lane) * 2;
    int f1 = ((kt * 2 + 1) * 32 + lane) * 2;
    acc0 = wmma_bf16(a, make_frag(bq[f0], bq[f0 + 1]), acc0);
    acc1 = wmma_bf16(a, make_frag(bq[f1], bq[f1 + 1]), acc1);
  }
  int colb = lane & 15;
  float bias0 = b10[colb], bias1 = b10[16 + colb];
#pragma unroll
  for (int v = 0; v < 8; ++v) {
    int row = rowbase + hi * 8 + v;
    float v0 = acc0[v] + bias0; v0 = v0 > 0.f ? v0 : 0.f;
    float v1 = acc1[v] + bias1; v1 = v1 > 0.f ? v1 : 0.f;
    u0[(size_t)row * 32 + colb]      = f2bf(v0);
    u0[(size_t)row * 32 + 16 + colb] = f2bf(v1);
  }
}

// ---------------------------------------------------------------------------
// Stage 2 (fused): t = relu(conv3(xb,W00)+b00), u1 = relu(conv3(u0,W11)+b11).
// 2 M-tiles per wave (32 rows) so each LDS B-fragment feeds 2 WMMAs. Weights
// double-buffered via async global->LDS; gather rows prefetched 1 offset ahead.
// Masked neighbors redirect to a zeroed dummy row -> contribution is 0.
// ---------------------------------------------------------------------------
__global__ __launch_bounds__(256) void k_conv_mid(const char* __restrict__ xb,
                                                  const char* __restrict__ u0,
                                                  const int* __restrict__ cidx,
                                                  const u32* __restrict__ w00p,
                                                  const u32* __restrict__ w11p,
                                                  const float* __restrict__ b00,
                                                  const float* __restrict__ b11,
                                                  u16* __restrict__ t,
                                                  u16* __restrict__ u1,
                                                  const char* __restrict__ zerorow) {
  __shared__ u32x4 shW00q[2][512];   // 2 x 8KB  (one offset of W00 packed)
  __shared__ u32x4 shW11q[2][128];   // 2 x 2KB  (one offset of W11 packed)

  int tid = threadIdx.x, lane = tid & 31, wave = tid >> 5, hi = lane >> 4;
  int rowbase = blockIdx.x * 256 + wave * 32;     // 2 M-tiles of 16 rows
  int mrow0 = rowbase + (lane & 15);
  int mrow1 = mrow0 + 16;

  auto stage = [&](int k, int buf) {
    const char* s00 = (const char*)(w00p + (size_t)k * 2048);
    const char* s11 = (const char*)(w11p + (size_t)k * 512);
#if HAVE_ASYNC_LDS
    char* d00 = (char*)shW00q[buf];
    char* d11 = (char*)shW11q[buf];
    async_copy16(s00 + tid * 16, d00 + tid * 16);
    async_copy16(s00 + 4096 + tid * 16, d00 + 4096 + tid * 16);
    if (tid < 128) async_copy16(s11 + tid * 16, d11 + tid * 16);
#else
    u32* d00 = (u32*)shW00q[buf];
    u32* d11 = (u32*)shW11q[buf];
    const u32* g00 = (const u32*)s00;
    const u32* g11 = (const u32*)s11;
#pragma unroll
    for (int r = 0; r < 8; ++r) d00[tid + r * 256] = g00[tid + r * 256];
    d11[tid] = g11[tid];
    d11[tid + 256] = g11[tid + 256];
#endif
  };

  v8f accT[2][2], accU[2][2];
#pragma unroll
  for (int m = 0; m < 2; ++m)
#pragma unroll
    for (int nt = 0; nt < 2; ++nt) {
      accT[m][nt] = (v8f){0,0,0,0,0,0,0,0};
      accU[m][nt] = (v8f){0,0,0,0,0,0,0,0};
    }

  stage(0, 0);
  int idx0 = cidx[mrow0];
  int idx1 = cidx[mrow1];
  stage_fence();

  for (int k = 0; k < KOFF; ++k) {
    int buf = k & 1;
    if (k + 1 < KOFF) stage(k + 1, buf ^ 1);

    const char* xr[2];
    const char* ur[2];
    xr[0] = (idx0 >= 0) ? xb + (size_t)idx0 * 256 : zerorow;
    ur[0] = (idx0 >= 0) ? u0 + (size_t)idx0 * 64  : zerorow;
    xr[1] = (idx1 >= 0) ? xb + (size_t)idx1 * 256 : zerorow;
    ur[1] = (idx1 >= 0) ? u0 + (size_t)idx1 * 64  : zerorow;

    // Pipeline next offset's kernel-map indices and prefetch gather rows.
    if (k + 1 < KOFF) {
      idx0 = cidx[(k + 1) * NVOX + mrow0];
      idx1 = cidx[(k + 1) * NVOX + mrow1];
      const char* p0 = (idx0 >= 0) ? xb + (size_t)idx0 * 256 : zerorow;
      const char* p1 = (idx1 >= 0) ? xb + (size_t)idx1 * 256 : zerorow;
      __builtin_prefetch(p0, 0, 0);
      __builtin_prefetch(p0 + 128, 0, 0);
      __builtin_prefetch(p1, 0, 0);
      __builtin_prefetch(p1 + 128, 0, 0);
      __builtin_prefetch((idx0 >= 0) ? u0 + (size_t)idx0 * 64 : zerorow, 0, 0);
      __builtin_prefetch((idx1 >= 0) ? u0 + (size_t)idx1 * 64 : zerorow, 0, 0);
    }

    // branch1 conv (K=32, one kt, two nt) -- B fragments reused across M-tiles
    {
      int f0 = (0 * 32 + lane) * 2, f1 = (1 * 32 + lane) * 2;
      v16bf b0 = make_frag(shW11q[buf][f0], shW11q[buf][f0 + 1]);
      v16bf b1 = make_frag(shW11q[buf][f1], shW11q[buf][f1 + 1]);
#pragma unroll
      for (int m = 0; m < 2; ++m) {
        u32x4 alo = *(const u32x4*)(ur[m] + hi * 16);
        u32x4 ahi = *(const u32x4*)(ur[m] + hi * 16 + 32);
        v16bf a = make_frag(alo, ahi);
        accU[m][0] = wmma_bf16(a, b0, accU[m][0]);
        accU[m][1] = wmma_bf16(a, b1, accU[m][1]);
      }
    }
    // branch0 conv (K=128: 4 kt, two nt)
#pragma unroll
    for (int kt = 0; kt < 4; ++kt) {
      int f0 = ((kt * 2 + 0) * 32 + lane) * 2;
      int f1 = ((kt * 2 + 1) * 32 + lane) * 2;
      v16bf b0 = make_frag(shW00q[buf][f0], shW00q[buf][f0 + 1]);
      v16bf b1 = make_frag(shW00q[buf][f1], shW00q[buf][f1 + 1]);
#pragma unroll
      for (int m = 0; m < 2; ++m) {
        u32x4 alo = *(const u32x4*)(xr[m] + kt * 64 + hi * 16);
        u32x4 ahi = *(const u32x4*)(xr[m] + kt * 64 + hi * 16 + 32);
        v16bf a = make_frag(alo, ahi);
        accT[m][0] = wmma_bf16(a, b0, accT[m][0]);
        accT[m][1] = wmma_bf16(a, b1, accT[m][1]);
      }
    }

    if (k + 1 < KOFF) stage_fence();
  }

  int colb = lane & 15;
  float bT0 = b00[colb], bT1 = b00[16 + colb];
  float bU0 = b11[colb], bU1 = b11[16 + colb];
#pragma unroll
  for (int m = 0; m < 2; ++m) {
#pragma unroll
    for (int v = 0; v < 8; ++v) {
      int row = rowbase + m * 16 + hi * 8 + v;
      float t0 = accT[m][0][v] + bT0; t0 = t0 > 0.f ? t0 : 0.f;
      float t1 = accT[m][1][v] + bT1; t1 = t1 > 0.f ? t1 : 0.f;
      float w0 = accU[m][0][v] + bU0; w0 = w0 > 0.f ? w0 : 0.f;
      float w1 = accU[m][1][v] + bU1; w1 = w1 > 0.f ? w1 : 0.f;
      t[(size_t)row * 32 + colb]       = f2bf(t0);
      t[(size_t)row * 32 + 16 + colb]  = f2bf(t1);
      u1[(size_t)row * 32 + colb]      = f2bf(w0);
      u1[(size_t)row * 32 + 16 + colb] = f2bf(w1);
    }
  }
}

// ---------------------------------------------------------------------------
// Stage 3 (fused): out0 = conv3(t,W01)+b01, out1 = u1@W12+b12,
//                  out = concat(out0,out1) + x   (f32 output [N,128]).
// ---------------------------------------------------------------------------
__global__ __launch_bounds__(256) void k_final(const char* __restrict__ t,
                                               const char* __restrict__ u1,
                                               const int* __restrict__ cidx,
                                               const u32* __restrict__ w01p,
                                               const u32* __restrict__ w12p,
                                               const float* __restrict__ b01,
                                               const float* __restrict__ b12,
                                               const float* __restrict__ x,
                                               float* __restrict__ out,
                                               const char* __restrict__ zerorow) {
  __shared__ u32x4 shW01q[2][256];   // 2 x 4KB (one offset of W01 packed)
  __shared__ u32x4 shW12q[256];      // 4KB W12 packed

  int tid = threadIdx.x, lane = tid & 31, wave = tid >> 5, hi = lane >> 4;
  int rowbase = blockIdx.x * 128 + wave * 16;
  int myrow = rowbase + (lane & 15);

  auto stage01 = [&](int k, int buf) {
    const char* s = (const char*)(w01p + (size_t)k * 1024);
#if HAVE_ASYNC_LDS
    char* d = (char*)shW01q[buf];
    async_copy16(s + tid * 16, d + tid * 16);
#else
    u32* d = (u32*)shW01q[buf];
    const u32* g = (const u32*)s;
#pragma unroll
    for (int r = 0; r < 4; ++r) d[tid + r * 256] = g[tid + r * 256];
#endif
  };

  // Stage W12 (once) + first offset of W01.
#if HAVE_ASYNC_LDS
  async_copy16((const char*)w12p + tid * 16, (char*)shW12q + tid * 16);
#else
  {
    u32* d = (u32*)shW12q;
#pragma unroll
    for (int r = 0; r < 4; ++r) d[tid + r * 256] = w12p[tid + r * 256];
  }
#endif
  stage01(0, 0);
  int idx = cidx[myrow];
  stage_fence();

  v8f acc[8];
#pragma unroll
  for (int i = 0; i < 8; ++i) acc[i] = (v8f){0,0,0,0,0,0,0,0};

  // out1 = u1 @ W12 (dense rows, K=32, 4 nt)
  {
    const char* urow = u1 + (size_t)myrow * 64;
    u32x4 alo = *(const u32x4*)(urow + hi * 16);
    u32x4 ahi = *(const u32x4*)(urow + hi * 16 + 32);
    v16bf a = make_frag(alo, ahi);
#pragma unroll
    for (int nt = 0; nt < 4; ++nt) {
      int f = (nt * 32 + lane) * 2;
      acc[4 + nt] = wmma_bf16(a, make_frag(shW12q[f], shW12q[f + 1]), acc[4 + nt]);
    }
  }

  // out0 = conv3(t, W01) (gather, K=32, 4 nt, 27 offsets)
  for (int k = 0; k < KOFF; ++k) {
    int buf = k & 1;
    if (k + 1 < KOFF) stage01(k + 1, buf ^ 1);

    const char* trow = (idx >= 0) ? t + (size_t)idx * 64 : zerorow;
    if (k + 1 < KOFF) {
      idx = cidx[(k + 1) * NVOX + myrow];
      __builtin_prefetch((idx >= 0) ? t + (size_t)idx * 64 : zerorow, 0, 0);
    }

    u32x4 alo = *(const u32x4*)(trow + hi * 16);
    u32x4 ahi = *(const u32x4*)(trow + hi * 16 + 32);
    v16bf a = make_frag(alo, ahi);
#pragma unroll
    for (int nt = 0; nt < 4; ++nt) {
      int f = (nt * 32 + lane) * 2;
      acc[nt] = wmma_bf16(a, make_frag(shW01q[buf][f], shW01q[buf][f + 1]), acc[nt]);
    }

    if (k + 1 < KOFF) stage_fence();
  }

  int colb = lane & 15;
#pragma unroll
  for (int v = 0; v < 8; ++v) {
    int row = rowbase + hi * 8 + v;
    const float* xr = x + (size_t)row * 128;
    float* orow = out + (size_t)row * 128;
#pragma unroll
    for (int nt = 0; nt < 4; ++nt) {
      int c = nt * 16 + colb;
      orow[c] = acc[nt][v] + b01[c] + xr[c];
      orow[64 + c] = acc[4 + nt][v] + b12[c] + xr[64 + c];
    }
  }
}

// ---------------------------------------------------------------------------
// Launcher
// ---------------------------------------------------------------------------
extern "C" void kernel_launch(void* const* d_in, const int* in_sizes, int n_in,
                              void* d_out, int out_size, void* d_ws, size_t ws_size,
                              hipStream_t stream) {
  const float* x  = (const float*)d_in[0];
  const int* nbr  = (const int*)d_in[1];
  const unsigned char* mask = (const unsigned char*)d_in[2];
  const float* W00 = (const float*)d_in[3];
  const float* b00 = (const float*)d_in[4];
  const float* W01 = (const float*)d_in[5];
  const float* b01 = (const float*)d_in[6];
  const float* W10 = (const float*)d_in[7];
  const float* b10 = (const float*)d_in[8];
  const float* W11 = (const float*)d_in[9];
  const float* b11 = (const float*)d_in[10];
  const float* W12 = (const float*)d_in[11];
  const float* b12 = (const float*)d_in[12];
  float* out = (float*)d_out;

  char* ws = (char*)d_ws;
  size_t off = 0;
  auto alloc = [&](size_t bytes) {
    size_t o = off;
    off += (bytes + 255) & ~(size_t)255;
    return o;
  };
  size_t o_xb   = alloc((size_t)NVOX * 128 * 2);   // 67 MB bf16 features
  size_t o_u0   = alloc((size_t)NVOX * 32 * 2);
  size_t o_t    = alloc((size_t)NVOX * 32 * 2);
  size_t o_u1   = alloc((size_t)NVOX * 32 * 2);
  size_t o_cidx = alloc((size_t)KOFF * NVOX * 4);
  size_t o_zero = alloc(512);
  size_t o_w00p = alloc((size_t)KOFF * 2048 * 4);
  size_t o_w01p = alloc((size_t)KOFF * 1024 * 4);
  size_t o_w11p = alloc((size_t)KOFF * 512 * 4);
  size_t o_w12p = alloc((size_t)1024 * 4);
  size_t o_w10p = alloc((size_t)2048 * 4);

  char* xb   = ws + o_xb;
  u16*  u0   = (u16*)(ws + o_u0);
  u16*  t    = (u16*)(ws + o_t);
  u16*  u1   = (u16*)(ws + o_u1);
  int*  cidx = (int*)(ws + o_cidx);
  u32*  zero = (u32*)(ws + o_zero);
  u32*  w00p = (u32*)(ws + o_w00p);
  u32*  w01p = (u32*)(ws + o_w01p);
  u32*  w11p = (u32*)(ws + o_w11p);
  u32*  w12p = (u32*)(ws + o_w12p);
  u32*  w10p = (u32*)(ws + o_w10p);

  // Preprocess
  k_convert_x<<<4096, 256, 0, stream>>>((const float2*)x, (u32*)xb, zero);
  k_build_cidx<<<4096, 256, 0, stream>>>(nbr, mask, cidx);
  k_pack_b<<<216, 256, 0, stream>>>(W00, w00p, KOFF, 128, 32);
  k_pack_b<<<108, 256, 0, stream>>>(W01, w01p, KOFF, 32, 64);
  k_pack_b<<<54, 256, 0, stream>>>(W11, w11p, KOFF, 32, 32);
  k_pack_b<<<4, 256, 0, stream>>>(W12, w12p, 1, 32, 64);
  k_pack_b<<<8, 256, 0, stream>>>(W10, w10p, 1, 128, 32);

  k_u0<<<NVOX / 128, 256, 0, stream>>>(xb, w10p, b10, u0);
  k_conv_mid<<<NVOX / 256, 256, 0, stream>>>(xb, (const char*)u0, cidx, w00p,
                                             w11p, b00, b11, t, u1, ws + o_zero);
  k_final<<<NVOX / 128, 256, 0, stream>>>((const char*)t, (const char*)u1, cidx,
                                          w01p, w12p, b01, b12, x, out,
                                          ws + o_zero);
}
